// VanillaTransformer_26225070309970
// MI455X (gfx1250) — compile-verified, matrix operand order checked by reading, not run on hardware
//
#include <hip/hip_runtime.h>
#include <stdint.h>

#define S_LEN  4096
#define D_DIM  1024
#define A_DIM  1024
#define FF_DIM 4096
#define BATCH  4
#define M_ROWS (BATCH * S_LEN)   // 16384

typedef __attribute__((ext_vector_type(16))) __bf16 v16bf;
typedef __attribute__((ext_vector_type(2)))  __bf16 v2bf;
typedef __attribute__((ext_vector_type(8)))  float  v8f;
typedef __attribute__((ext_vector_type(2)))  float  v2f;

__device__ __forceinline__ unsigned pack_bf16x2(float a, float b) {
  v2f t; t[0] = a; t[1] = b;
  union { v2bf v; unsigned u; } r;
  r.v = __builtin_convertvector(t, v2bf);   // v_cvt_pk_bf16_f32
  return r.u;
}

__device__ __forceinline__ float gelu_f(float x) {
  const float c0 = 0.7978845608028654f;  // sqrt(2/pi)
  float t = tanhf(c0 * (x + 0.044715f * x * x * x));
  return 0.5f * x * (1.0f + t);
}

__device__ __forceinline__ float elu1_f(float x) {
  return x > 0.0f ? x + 1.0f : __expf(x);   // elu(x)+1
}

// C[M,N] = act(A[M,K] @ W[K,N] + bias[N]); fp32 in/out, bf16 WMMA compute.
// Block tile 256x128, wave tile 64x64 (4x4 WMMA), BK=32, double-buffered LDS.
// ACT: 0 = none, 1 = elu+1, 2 = gelu, 3 = gelu(gelu(x))
template<int ACT>
__global__ __launch_bounds__(256)
void gemm_bias_act(const float* __restrict__ A, const float* __restrict__ W,
                   const float* __restrict__ bias, float* __restrict__ C,
                   int M, int N, int K) {
  __shared__ __attribute__((aligned(16))) unsigned short sA[2][256][32]; // [m][k] bf16 bits, 32 KB
  __shared__ __attribute__((aligned(16))) unsigned short sB[2][128][32]; // [n][k] bf16 bits, 16 KB

  const int tid    = threadIdx.x;
  const int blockN = blockIdx.x * 128;
  const int blockM = blockIdx.y * 256;
  const int lane   = tid & 31;
  const int hlf    = lane >> 4;     // 16-lane half selects K chunk / +8 rows of C
  const int r16    = lane & 15;
  const int w      = tid >> 5;      // wave id 0..7
  const int wm     = (w & 3) * 64;  // 4 waves along M
  const int wn     = (w >> 2) * 64; // 2 waves along N

  v8f acc[4][4];
  v8f zv;
  #pragma unroll
  for (int e = 0; e < 8; ++e) zv[e] = 0.0f;
  #pragma unroll
  for (int i = 0; i < 4; ++i)
    #pragma unroll
    for (int j = 0; j < 4; ++j) acc[i][j] = zv;

  const int KT = K >> 5;            // number of 32-wide K tiles
  float4 ra[8], rb[2][2];           // A: 8 float4; B: 2 work items x (k, k+1) rows

  auto load_regs = [&](int kt) {
    const int k0 = kt << 5;
    #pragma unroll
    for (int it = 0; it < 8; ++it) {          // A: 256x32 fp32
      int idx = tid + it * 256;               // 0..2047 float4 slots
      int row = idx >> 3;                     // 8 float4 per 32-wide row
      int col = (idx & 7) * 4;
      ra[it] = *(const float4*)(A + (size_t)(blockM + row) * K + k0 + col);
    }
    #pragma unroll
    for (int it = 0; it < 2; ++it) {          // B: 32x128 fp32, 2 consecutive k rows/item
      int idx   = tid + it * 256;             // 0..511 work items
      int kpair = idx >> 5;                   // 16 k-pairs
      int col   = (idx & 31) * 4;
      const float* p = W + (size_t)(k0 + 2 * kpair) * N + blockN + col;
      rb[it][0] = *(const float4*)(p);        // row k
      rb[it][1] = *(const float4*)(p + N);    // row k+1
    }
  };

  auto store_lds = [&](int buf) {
    #pragma unroll
    for (int it = 0; it < 8; ++it) {
      int idx = tid + it * 256;
      int row = idx >> 3;
      int col = (idx & 7) * 4;
      uint2 u2;
      u2.x = pack_bf16x2(ra[it].x, ra[it].y);
      u2.y = pack_bf16x2(ra[it].z, ra[it].w);
      *(uint2*)&sA[buf][row][col] = u2;       // 8B ds_store
    }
    #pragma unroll
    for (int it = 0; it < 2; ++it) {          // transposed scatter into [n][k], packed k-pairs
      int idx   = tid + it * 256;
      int kpair = idx >> 5;
      int col   = (idx & 31) * 4;
      *(unsigned*)&sB[buf][col + 0][2 * kpair] = pack_bf16x2(rb[it][0].x, rb[it][1].x);
      *(unsigned*)&sB[buf][col + 1][2 * kpair] = pack_bf16x2(rb[it][0].y, rb[it][1].y);
      *(unsigned*)&sB[buf][col + 2][2 * kpair] = pack_bf16x2(rb[it][0].z, rb[it][1].z);
      *(unsigned*)&sB[buf][col + 3][2 * kpair] = pack_bf16x2(rb[it][0].w, rb[it][1].w);
    }
  };

  union FragU { v16bf v; float4 f[2]; };

  auto compute = [&](int buf) {
    FragU af[4], bfr[4];
    #pragma unroll
    for (int mt = 0; mt < 4; ++mt) {
      const unsigned short* p = &sA[buf][wm + mt * 16 + r16][0];
      af[mt].f[0] = *(const float4*)(p + hlf * 8);        // K = hlf*8 .. +7
      af[mt].f[1] = *(const float4*)(p + 16 + hlf * 8);   // K = 16+hlf*8 .. +7
    }
    #pragma unroll
    for (int nt = 0; nt < 4; ++nt) {
      const unsigned short* p = &sB[buf][wn + nt * 16 + r16][0];
      bfr[nt].f[0] = *(const float4*)(p + hlf * 8);
      bfr[nt].f[1] = *(const float4*)(p + 16 + hlf * 8);
    }
    #pragma unroll
    for (int mt = 0; mt < 4; ++mt)
      #pragma unroll
      for (int nt = 0; nt < 4; ++nt)
        acc[mt][nt] = __builtin_amdgcn_wmma_f32_16x16x32_bf16(
            false, af[mt].v, false, bfr[nt].v, (short)0, acc[mt][nt], false, false);
  };

  // software pipeline: prefetch k+1 while computing k; ping-pong LDS
  load_regs(0);
  store_lds(0);
  __syncthreads();
  for (int kt = 0; kt < KT; ++kt) {
    const int cur = kt & 1;
    const bool more = (kt + 1) < KT;
    if (more) load_regs(kt + 1);
    compute(cur);
    if (more) store_lds(cur ^ 1);
    __syncthreads();
  }

  // --- epilogue: bias + activation, fp32 store ---
  #pragma unroll
  for (int mt = 0; mt < 4; ++mt) {
    #pragma unroll
    for (int nt = 0; nt < 4; ++nt) {
      int col = blockN + wn + nt * 16 + r16;
      float b = bias[col];
      #pragma unroll
      for (int e = 0; e < 8; ++e) {
        int row = blockM + wm + mt * 16 + hlf * 8 + e;  // C layout: VGPR e -> M=e (+8 hi half)
        float v = acc[mt][nt][e] + b;
        if (ACT == 1)      v = elu1_f(v);
        else if (ACT == 2) v = gelu_f(v);
        else if (ACT == 3) v = gelu_f(gelu_f(v));
        C[(size_t)row * N + col] = v;
      }
    }
  }
}

__global__ void zero_kernel(float* p, int n) {
  int i = blockIdx.x * 256 + threadIdx.x;
  if (i < n) p[i] = 0.0f;
}

// KV[b,h] = sum_s K[b,s,h]*V[b,s,h];  Ksum[b,h] = sum_s K[b,s,h]
__global__ __launch_bounds__(256)
void kv_reduce_kernel(const float* __restrict__ Kf, const float* __restrict__ Vf,
                      float* __restrict__ KV, float* __restrict__ Ksum) {
  int b  = blockIdx.x;                       // BATCH
  int h  = blockIdx.y * 256 + threadIdx.x;   // A_DIM/256
  int sc = blockIdx.z;                       // S_LEN/128 chunks
  size_t base = ((size_t)b * S_LEN + (size_t)sc * 128) * A_DIM + h;
  float kv = 0.f, ks = 0.f;
  for (int s = 0; s < 128; ++s) {
    float kk = Kf[base + (size_t)s * A_DIM];
    float vv = Vf[base + (size_t)s * A_DIM];
    kv = fmaf(kk, vv, kv);
    ks += kk;
  }
  atomicAdd(&KV[b * A_DIM + h], kv);
  atomicAdd(&Ksum[b * A_DIM + h], ks);
}

// In place: Q -> Q * KV[b,:] / (Q * Ksum[b,:] + 1e-6)
__global__ __launch_bounds__(256)
void attn_v_kernel(float* __restrict__ Q, const float* __restrict__ KV,
                   const float* __restrict__ Ksum) {
  size_t i = (size_t)blockIdx.x * 256 + threadIdx.x;   // over B*S*A = 2^24
  int h = (int)(i & (A_DIM - 1));
  int b = (int)(i >> 22);                              // / (S*A)
  float q = Q[i];
  float z = 1.0f / (q * Ksum[b * A_DIM + h] + 1e-6f);
  Q[i] = q * KV[b * A_DIM + h] * z;
}

// out = LN(a + res) * sc + bi, rows of D_DIM
__global__ __launch_bounds__(256)
void add_ln_kernel(const float* __restrict__ a, const float* __restrict__ res,
                   const float* __restrict__ sc, const float* __restrict__ bi,
                   float* __restrict__ out) {
  __shared__ float sh1[256], sh2[256];
  int row = blockIdx.x;
  int t = threadIdx.x;
  const float* pa = a + (size_t)row * D_DIM;
  const float* pr = res + (size_t)row * D_DIM;
  float v[4]; float s1 = 0.f, s2 = 0.f;
  #pragma unroll
  for (int j = 0; j < 4; ++j) {
    int c = t + j * 256;
    v[j] = pa[c] + pr[c];
    s1 += v[j]; s2 += v[j] * v[j];
  }
  sh1[t] = s1; sh2[t] = s2;
  __syncthreads();
  for (int off = 128; off > 0; off >>= 1) {
    if (t < off) { sh1[t] += sh1[t + off]; sh2[t] += sh2[t + off]; }
    __syncthreads();
  }
  float mean = sh1[0] * (1.0f / D_DIM);
  float var  = sh2[0] * (1.0f / D_DIM) - mean * mean;
  float rstd = rsqrtf(var + 1e-6f);
  #pragma unroll
  for (int j = 0; j < 4; ++j) {
    int c = t + j * 256;
    out[(size_t)row * D_DIM + c] = (v[j] - mean) * rstd * sc[c] + bi[c];
  }
}

extern "C" void kernel_launch(void* const* d_in, const int* in_sizes, int n_in,
                              void* d_out, int out_size, void* d_ws, size_t ws_size,
                              hipStream_t stream) {
  const float* x     = (const float*)d_in[0];
  const float* Wq    = (const float*)d_in[1];
  const float* bq    = (const float*)d_in[2];
  const float* Wk    = (const float*)d_in[3];
  const float* bk    = (const float*)d_in[4];
  const float* Wv    = (const float*)d_in[5];
  const float* bv    = (const float*)d_in[6];
  const float* Wo    = (const float*)d_in[7];
  const float* bo    = (const float*)d_in[8];
  const float* ln1s  = (const float*)d_in[9];
  const float* ln1b  = (const float*)d_in[10];
  const float* W1    = (const float*)d_in[11];
  const float* b1    = (const float*)d_in[12];
  const float* W2    = (const float*)d_in[13];
  const float* b2    = (const float*)d_in[14];
  const float* ln2s  = (const float*)d_in[15];
  const float* ln2b  = (const float*)d_in[16];

  char* ws = (char*)d_ws;
  const size_t XB = (size_t)M_ROWS * D_DIM * sizeof(float);   // 64 MB
  float* buf_x  = (float*)(ws + 0 * XB);
  float* buf_q  = (float*)(ws + 1 * XB);   // Q, then V after attn_v
  float* buf_k  = (float*)(ws + 2 * XB);
  float* buf_v  = (float*)(ws + 3 * XB);
  float* buf_a  = (float*)(ws + 4 * XB);   // attn out, later FFN out
  float* buf_x2 = (float*)(ws + 5 * XB);   // post-LN1 (FFN residual)
  float* buf_h  = (float*)(ws + 6 * XB);   // 16384x4096 = 256 MB
  float* kvbuf  = (float*)(ws + 6 * XB + (size_t)M_ROWS * FF_DIM * sizeof(float));
  float* ksbuf  = kvbuf + BATCH * A_DIM;

  hipMemcpyAsync(buf_x, x, XB, hipMemcpyDeviceToDevice, stream);

  dim3 blk(256);
  dim3 gN1(A_DIM  / 128, M_ROWS / 256);   // N = 1024 outputs
  dim3 gN4(FF_DIM / 128, M_ROWS / 256);   // N = 4096 outputs

  for (int i = 0; i < 4; ++i) {
    const float* Wqi = Wq + (size_t)i * D_DIM * A_DIM;
    const float* Wki = Wk + (size_t)i * D_DIM * A_DIM;
    const float* Wvi = Wv + (size_t)i * D_DIM * A_DIM;
    const float* Woi = Wo + (size_t)i * A_DIM * D_DIM;
    const float* W1i = W1 + (size_t)i * D_DIM * FF_DIM;
    const float* W2i = W2 + (size_t)i * FF_DIM * D_DIM;
    const float* bqi = bq + (size_t)i * A_DIM;
    const float* bki = bk + (size_t)i * A_DIM;
    const float* bvi = bv + (size_t)i * A_DIM;
    const float* boi = bo + (size_t)i * D_DIM;
    const float* b1i = b1 + (size_t)i * FF_DIM;
    const float* b2i = b2 + (size_t)i * D_DIM;

    // Q = elu(x@Wq+bq)+1 ; K = elu(x@Wk+bk)+1 ; v = x@Wv+bv
    gemm_bias_act<1><<<gN1, blk, 0, stream>>>(buf_x, Wqi, bqi, buf_q, M_ROWS, A_DIM, D_DIM);
    gemm_bias_act<1><<<gN1, blk, 0, stream>>>(buf_x, Wki, bki, buf_k, M_ROWS, A_DIM, D_DIM);
    gemm_bias_act<0><<<gN1, blk, 0, stream>>>(buf_x, Wvi, bvi, buf_v, M_ROWS, A_DIM, D_DIM);

    // KV / Ksum aggregation over S
    zero_kernel<<<(2 * BATCH * A_DIM + 255) / 256, blk, 0, stream>>>(kvbuf, 2 * BATCH * A_DIM);
    kv_reduce_kernel<<<dim3(BATCH, A_DIM / 256, S_LEN / 128), blk, 0, stream>>>(buf_k, buf_v, kvbuf, ksbuf);

    // V = Q*KV / (Q*Ksum + 1e-6) (in place on buf_q)
    attn_v_kernel<<<(M_ROWS * A_DIM) / 256, blk, 0, stream>>>(buf_q, kvbuf, ksbuf);

    // a = gelu(gelu(V@Wo + bo)) ; x2 = LN(a + x)
    gemm_bias_act<3><<<gN1, blk, 0, stream>>>(buf_q, Woi, boi, buf_a, M_ROWS, D_DIM, A_DIM);
    add_ln_kernel<<<M_ROWS, blk, 0, stream>>>(buf_a, buf_x, ln1s + (size_t)i * D_DIM,
                                              ln1b + (size_t)i * D_DIM, buf_x2);

    // FFN: h = gelu(x2@W1+b1) ; f = gelu(h@W2+b2) ; x = LN(f + x2)
    gemm_bias_act<2><<<gN4, blk, 0, stream>>>(buf_x2, W1i, b1i, buf_h, M_ROWS, FF_DIM, D_DIM);
    gemm_bias_act<2><<<gN1, blk, 0, stream>>>(buf_h, W2i, b2i, buf_a, M_ROWS, D_DIM, FF_DIM);
    float* xo = (i == 3) ? (float*)d_out : buf_x;
    add_ln_kernel<<<M_ROWS, blk, 0, stream>>>(buf_a, buf_x2, ln2s + (size_t)i * D_DIM,
                                              ln2b + (size_t)i * D_DIM, xo);
  }
  (void)in_sizes; (void)n_in; (void)out_size; (void)ws_size;
}
